// BronxLayer_67619965108966
// MI455X (gfx1250) — matrix operand order repaired and checked by reading, not compiled, not run on hardware
//
#include <hip/hip_runtime.h>
#include <hip/hip_bf16.h>
#include <stdint.h>

// ---------------------------------------------------------------------------
// BronxLayer fused kernels for gfx1250 (MI455X), wave32 + WMMA bf16 16x16x32.
//   Pass A (prep):   LayerNorm + k/mu/ls projections -> bf16 workspace
//                    (k pre-scaled by 128^-0.5 * log2(e): scores come out of
//                     the WMMA already in the base-2 domain of v_exp/v_log)
//   Pass A2(repack): swizzle hn into WMMA-B-fragment-ready layout
//   Pass B (main):   fused scores (WMMA, transposed) -> softplus/exp/KL with
//                    raw v_exp_f32/v_log_f32 -> aggregation (WMMA chain,
//                    no cross-lane shuffles) -> LDS reduce
//   Pass C (final):  L1-normalize, w_v projection, ELU, KL scalar
// ---------------------------------------------------------------------------

typedef __bf16 bf16_t;
typedef __attribute__((ext_vector_type(16))) __bf16 v16bf;
typedef __attribute__((ext_vector_type(8)))  float  v8f;

#define NN   4096
#define HH   128
#define DHD  64
#define KSCALE   0.08838834764831845f    // 128^-0.5
#define L2E      1.4426950408889634f     // log2(e)
#define LN2      0.6931471805599453f     // ln(2)
#define KPRE     (KSCALE * L2E)          // folded into k at prep time
#define SPTHR    28.853900817779268f     // 20 * log2(e)
#define KL_C1    0.24022650695910072f    // 0.5 * ln2^2
#define KL_C3   -0.13348707941833565f    // -ln(ln2) - 0.5

struct alignas(16) FragBits { uint4 a, b; };

// Build a v16bf fragment from two contiguous 16-byte runs (-> 2x b128 loads).
__device__ __forceinline__ v16bf load_frag2(const bf16_t* lo, const bf16_t* hi) {
    FragBits p;
    p.a = *(const uint4*)lo;
    p.b = *(const uint4*)hi;
    return __builtin_bit_cast(v16bf, p);
}

// ---------------------------------------------------------------------------
// Pass A: LayerNorm + projections. 256 blocks x 128 threads, 16 rows/block.
// ---------------------------------------------------------------------------
__global__ __launch_bounds__(128) void bronx_prep(
    const float* __restrict__ hin, const float* __restrict__ w_k,
    const float* __restrict__ w_mu, const float* __restrict__ w_ls,
    const float* __restrict__ gamma, const float* __restrict__ beta,
    bf16_t* __restrict__ hnb, bf16_t* __restrict__ kb,
    bf16_t* __restrict__ mub, bf16_t* __restrict__ lsb,
    float* __restrict__ kl_g)
{
    const int j = threadIdx.x;          // feature 0..127
    __shared__ float s_hn[HH];
    __shared__ float s_red[HH];

    if (blockIdx.x == 0 && j == 0) *kl_g = 0.f;   // zero KL accumulator

    for (int r = 0; r < 16; ++r) {
        const int n = blockIdx.x * 16 + r;
        const float v = hin[(size_t)n * HH + j];

        s_red[j] = v; __syncthreads();
        #pragma unroll
        for (int s = 64; s > 0; s >>= 1) { if (j < s) s_red[j] += s_red[j + s]; __syncthreads(); }
        const float mean = s_red[0] * (1.f / HH);
        __syncthreads();

        const float d = v - mean;
        s_red[j] = d * d; __syncthreads();
        #pragma unroll
        for (int s = 64; s > 0; s >>= 1) { if (j < s) s_red[j] += s_red[j + s]; __syncthreads(); }
        const float var = s_red[0] * (1.f / HH);
        __syncthreads();

        const float hn = d * rsqrtf(var + 1e-5f) * gamma[j] + beta[j];
        s_hn[j] = hn;
        hnb[(size_t)n * HH + j] = (bf16_t)hn;
        __syncthreads();

        float ak = 0.f, am = 0.f, al = 0.f;
        #pragma unroll 8
        for (int i = 0; i < HH; ++i) {
            const float hv = s_hn[i];
            ak = fmaf(hv, w_k [j * HH + i], ak);
            am = fmaf(hv, w_mu[j * HH + i], am);
            al = fmaf(hv, w_ls[j * HH + i], al);
        }
        // reshape(N, DH, HEADS): col j -> (y = j>>1, head = j&1); store [head][n][y]
        const int y = j >> 1, b = j & 1;
        const size_t o = ((size_t)b * NN + n) * DHD + y;
        kb[o]  = (bf16_t)(ak * KPRE);   // fold score scale + log2(e) into k
        mub[o] = (bf16_t)am;
        lsb[o] = (bf16_t)al;
        __syncthreads();
    }
}

// ---------------------------------------------------------------------------
// Pass A2: swizzle hn into B-fragment-ready layout:
//   hnf[((chunk*8 + nt)*32 + lane)*16 + h] = hn[chunk*32 + (h&7) + 8*(lane>>4) + 16*(h>>3)][nt*16 + (lane&15)]
// One thread per (chunk, nt, lane) = 32768 threads.
// ---------------------------------------------------------------------------
__global__ __launch_bounds__(256) void bronx_repack(
    const bf16_t* __restrict__ hnb, bf16_t* __restrict__ hnf)
{
    const int t    = blockIdx.x * 256 + threadIdx.x;  // 0..32767
    const int lane = t & 31;
    const int nt   = (t >> 5) & 7;
    const int zc   = (t >> 8) * 32;
    const int lm   = lane & 15, lg = lane >> 4;

    alignas(16) bf16_t tmp[16];
    #pragma unroll
    for (int hh = 0; hh < 16; ++hh)
        tmp[hh] = hnb[(size_t)(zc + (hh & 7) + 8 * lg + 16 * (hh >> 3)) * HH + nt * 16 + lm];

    uint4* dst = (uint4*)(hnf + (size_t)t * 16);
    dst[0] = ((const uint4*)tmp)[0];
    dst[1] = ((const uint4*)tmp)[1];
}

// ---------------------------------------------------------------------------
// Pass B: fused main kernel. 256 blocks (16 x-rows each) x 256 threads.
// wave w (0..7): head = w&1, 4 waves/head stride the 128 z-chunks of 32.
// ---------------------------------------------------------------------------
__global__ __launch_bounds__(256) void bronx_main(
    const float* __restrict__ eps, const float* __restrict__ diff,
    const bf16_t* __restrict__ hnf, const bf16_t* __restrict__ kb,
    const bf16_t* __restrict__ mub, const bf16_t* __restrict__ lsb,
    float* __restrict__ msg_un, float* __restrict__ den_g,
    float* __restrict__ kl_g)
{
    const int x0   = blockIdx.x * 16;
    const int lane = threadIdx.x & 31;
    const int wave = threadIdx.x >> 5;
    const int b    = wave & 1;          // head
    const int ws4  = wave >> 1;         // 0..3 slot within head
    const int lm   = lane & 15;
    const int lg   = lane >> 4;

    __shared__ float s_out[2][16][HH];  // un-normalized messages, both heads
    __shared__ float s_den[2][16];
    __shared__ float s_kl;

    for (int i = threadIdx.x; i < 2 * 16 * HH; i += 256) ((float*)s_out)[i] = 0.f;
    if (threadIdx.x < 32) ((float*)s_den)[threadIdx.x] = 0.f;
    if (threadIdx.x == 0) s_kl = 0.f;
    __syncthreads();

    const int xr = x0 + lm;             // this lane's x row (score & ew phases)

    // Hoisted B-fragments of k: B[K=y][N=x] = k[x,y,b]; h<8 -> K = h+8g ; h>=8 -> K = 16+(h-8)+8g
    const bf16_t* krow = kb + ((size_t)b * NN + xr) * DHD + 8 * lg;
    const v16bf kf0 = load_frag2(krow,      krow + 16);
    const v16bf kf1 = load_frag2(krow + 32, krow + 48);

    v8f acc[8];
    #pragma unroll
    for (int nt = 0; nt < 8; ++nt) acc[nt] = {};
    float den_acc = 0.f, kl_acc = 0.f;

    for (int c = ws4; c < 128; c += 4) {
        const int zc = c * 32;

        // prefetch next chunk of the two DRAM streams
        if (c + 4 < 128) {
            __builtin_prefetch(eps  + (((size_t)xr * NN + zc + 128) * 2 + b), 0, 1);
            __builtin_prefetch(diff + ((size_t)xr * NN + zc + 128), 0, 1);
        }

        // ---- transposed score tiles: D[M=z][N=x] = mu/ls (A) x k_scaled (B)
        //      (results are mu_s*log2e and ls_s*log2e: base-2 domain) ----
        v8f muT[2], lsT[2];
        #pragma unroll
        for (int zt = 0; zt < 2; ++zt) {
            const int zrow = zc + zt * 16 + lm;              // A M-row = z
            const bf16_t* mrow = mub + ((size_t)b * NN + zrow) * DHD + 8 * lg;
            const bf16_t* lrow = lsb + ((size_t)b * NN + zrow) * DHD + 8 * lg;
            // A-frag halves: h<8 -> K = h+8g ; h>=8 -> K = 16+(h-8)+8g  (2x16B runs)
            const v16bf amu0 = load_frag2(mrow,      mrow + 16);
            const v16bf amu1 = load_frag2(mrow + 32, mrow + 48);
            const v16bf als0 = load_frag2(lrow,      lrow + 16);
            const v16bf als1 = load_frag2(lrow + 32, lrow + 48);
            v8f cm = {};
            cm = __builtin_amdgcn_wmma_f32_16x16x32_bf16(false, amu0, false, kf0, (short)0, cm, false, false);
            cm = __builtin_amdgcn_wmma_f32_16x16x32_bf16(false, amu1, false, kf1, (short)0, cm, false, false);
            muT[zt] = cm;
            v8f cl = {};
            cl = __builtin_amdgcn_wmma_f32_16x16x32_bf16(false, als0, false, kf0, (short)0, cl, false, false);
            cl = __builtin_amdgcn_wmma_f32_16x16x32_bf16(false, als1, false, kf1, (short)0, cl, false, false);
            lsT[zt] = cl;
        }

        // ---- per-lane elementwise in base-2 domain; score C-layout IS the
        //      next WMMA's A-frag layout ----
        v16bf afrag;
        #pragma unroll
        for (int zt = 0; zt < 2; ++zt) {
            const int zbase = zc + zt * 16 + 8 * lg;
            // diffusion: 8 consecutive floats per lane -> 2x b128
            alignas(16) float dbuf[8];
            const float4* dp = (const float4*)(diff + (size_t)xr * NN + zbase);
            *(float4*)(&dbuf[0]) = dp[0];
            *(float4*)(&dbuf[4]) = dp[1];
            #pragma unroll
            for (int r = 0; r < 8; ++r) {
                const int z = zbase + r;                     // matches A K-pattern
                const float ms2 = muT[zt][r];                // mu_s * log2e
                const float ql2 = lsT[zt][r];                // ls_s * log2e
                // softplus * log2e: sg2 = log2(1 + 2^ql2)  (= sigma * log2e)
                const float u   = __builtin_amdgcn_exp2f(ql2);
                const float sg2 = (ql2 > SPTHR) ? ql2
                                  : __builtin_amdgcn_logf(1.f + u);
                const float dv  = dbuf[r];
                // KL: 0.5*(sg^2+ms^2) - ln(sg) - 0.5 with sg = ln2*sg2, ms = ln2*ms2
                const float klt = fmaf(fmaf(sg2, sg2, ms2 * ms2), KL_C1,
                                       fmaf(-LN2, __builtin_amdgcn_logf(sg2), KL_C3));
                kl_acc += (dv > 0.f) ? klt : 0.f;            // diff ~ U[0,1): sign in {0,1}
                const float ev = eps[((size_t)xr * NN + z) * 2 + b];
                // exp(mu + sigma*eps) = 2^(ms2 + sg2*eps)
                const float av = __builtin_amdgcn_exp2f(fmaf(sg2, ev, ms2)) * dv;
                den_acc += fabsf(av);
                afrag[zt * 8 + r] = (bf16_t)av;
            }
        }

        // ---- aggregation: out[x, z'] += a[x, z] * hn[z, z'] (8 N-tiles) ----
        #pragma unroll
        for (int nt = 0; nt < 8; ++nt) {
            const bf16_t* hrow = hnf + ((size_t)(c * 8 + nt) * 32 + lane) * 16;
            const v16bf hf = load_frag2(hrow, hrow + 8);     // one 32B run per lane
            acc[nt] = __builtin_amdgcn_wmma_f32_16x16x32_bf16(false, afrag, false, hf, (short)0, acc[nt], false, false);
        }
    }

    // ---- cross-wave reduction via LDS atomics ----
    // aggregation D layout: M = x = r + 8*lg, N = z' = nt*16 + lm
    #pragma unroll
    for (int nt = 0; nt < 8; ++nt) {
        #pragma unroll
        for (int r = 0; r < 8; ++r)
            atomicAdd(&s_out[b][r + 8 * lg][nt * 16 + lm], acc[nt][r]);
    }
    atomicAdd(&s_den[b][lm], den_acc);
    atomicAdd(&s_kl, kl_acc);
    __syncthreads();

    for (int i = threadIdx.x; i < 2 * 16 * HH; i += 256) {
        const int bb = i >> 11, r = (i >> 7) & 15, col = i & 127;
        msg_un[((size_t)bb * NN + x0 + r) * HH + col] = s_out[bb][r][col];
    }
    if (threadIdx.x < 32)
        den_g[(size_t)(threadIdx.x >> 4) * NN + x0 + (threadIdx.x & 15)] =
            s_den[threadIdx.x >> 4][threadIdx.x & 15];
    if (threadIdx.x == 0) atomicAdd(kl_g, s_kl);
}

// ---------------------------------------------------------------------------
// Pass C: L1-normalize + w_v projection + ELU + KL scalar. 4096 x 128.
// ---------------------------------------------------------------------------
__global__ __launch_bounds__(128) void bronx_final(
    const float* __restrict__ msg_un, const float* __restrict__ den_g,
    const float* __restrict__ w_v, const float* __restrict__ b_v,
    const float* __restrict__ kl_g, float* __restrict__ out)
{
    const int x = blockIdx.x, j = threadIdx.x;
    __shared__ float s_m[256];
    for (int cc = j; cc < 256; cc += 128) {
        const int bb = cc >> 7, zp = cc & 127;
        const float dn = fmaxf(den_g[(size_t)bb * NN + x], 1e-12f);
        s_m[cc] = msg_un[((size_t)bb * NN + x) * HH + zp] / dn;
    }
    __syncthreads();
    float a = b_v[j];
    #pragma unroll 8
    for (int cc = 0; cc < 256; ++cc) a = fmaf(s_m[cc], w_v[j * 256 + cc], a);
    out[(size_t)x * HH + j] = (a > 0.f) ? a : expm1f(a);    // ELU
    if (x == 0 && j == 0) out[(size_t)NN * HH] = kl_g[0] * (1.f / ((float)NN * (float)NN));
}

// ---------------------------------------------------------------------------
extern "C" void kernel_launch(void* const* d_in, const int* in_sizes, int n_in,
                              void* d_out, int out_size, void* d_ws, size_t ws_size,
                              hipStream_t stream) {
    (void)in_sizes; (void)n_in; (void)out_size; (void)ws_size;
    const float* hin   = (const float*)d_in[0];
    const float* diff  = (const float*)d_in[1];
    const float* eps   = (const float*)d_in[2];
    const float* w_k   = (const float*)d_in[3];
    const float* w_mu  = (const float*)d_in[4];
    const float* w_ls  = (const float*)d_in[5];
    const float* w_v   = (const float*)d_in[6];
    const float* b_v   = (const float*)d_in[7];
    const float* gamma = (const float*)d_in[8];
    const float* beta  = (const float*)d_in[9];
    float* out = (float*)d_out;

    char* w = (char*)d_ws;
    bf16_t* hnb = (bf16_t*)w; w += (size_t)NN * HH * sizeof(bf16_t);       // 1 MB
    bf16_t* hnf = (bf16_t*)w; w += (size_t)NN * HH * sizeof(bf16_t);       // 1 MB (swizzled)
    bf16_t* kb  = (bf16_t*)w; w += (size_t)2 * NN * DHD * sizeof(bf16_t);  // 1 MB
    bf16_t* mub = (bf16_t*)w; w += (size_t)2 * NN * DHD * sizeof(bf16_t);  // 1 MB
    bf16_t* lsb = (bf16_t*)w; w += (size_t)2 * NN * DHD * sizeof(bf16_t);  // 1 MB
    float* msg  = (float*)w;  w += (size_t)2 * NN * HH * sizeof(float);    // 4 MB
    float* den  = (float*)w;  w += (size_t)2 * NN * sizeof(float);         // 32 KB
    float* klg  = (float*)w;

    bronx_prep<<<dim3(NN / 16), dim3(128), 0, stream>>>(
        hin, w_k, w_mu, w_ls, gamma, beta, hnb, kb, mub, lsb, klg);
    bronx_repack<<<dim3(128), dim3(256), 0, stream>>>(hnb, hnf);
    bronx_main<<<dim3(NN / 16), dim3(256), 0, stream>>>(
        eps, diff, hnf, kb, mub, lsb, msg, den, klg);
    bronx_final<<<dim3(NN), dim3(128), 0, stream>>>(
        msg, den, w_v, b_v, klg, out);
}
